// BidirectionalSpatialMamba_506806140960
// MI455X (gfx1250) — compile-verified, matrix operand order checked by reading, not run on hardware
//
#include <hip/hip_runtime.h>
#include <math.h>

// ---------------------------------------------------------------------------
// Bidirectional spatial Mamba for MI455X (gfx1250, wave32, WMMA, async-LDS)
// ---------------------------------------------------------------------------

typedef __bf16 bf16;
typedef __attribute__((ext_vector_type(16))) __bf16 v16bf;
typedef __attribute__((ext_vector_type(8)))  float  v8f;

#define DEV __device__ __forceinline__

constexpr int kCH     = 96;
constexpr int kDSTATE = 16;
constexpr int kDIN    = 192;
constexpr int kDTR    = 6;
constexpr int kDBL    = kDTR + 2 * kDSTATE;   // 38
constexpr int kB      = 2;
constexpr int kL      = 128 * 128;            // 16384
constexpr int kNROW   = kB * kL;              // 32768
constexpr int kNC     = 64;                   // scan chunks
constexpr int kCL     = kL / kNC;             // 256 steps per chunk
constexpr int kTS     = 16;                   // scan LDS tile (steps)
constexpr int kNT     = kCL / kTS;            // tiles per chunk

// float -> bf16, round-to-nearest-even
DEV bf16 f2bf(float f) {
    unsigned int u = __float_as_uint(f);
    unsigned int r = (u + 0x7FFFu + ((u >> 16) & 1u)) >> 16;
    unsigned short s = (unsigned short)r;
    bf16 b;
    __builtin_memcpy(&b, &s, 2);
    return b;
}

DEV float silu(float v) { return v / (1.f + __expf(-v)); }

// Async global->LDS copy (CDNA5 GLOBAL_LOAD_ASYNC_TO_LDS_B32, ASYNCcnt).
// LDS destination address = low 32 bits of the generic pointer (LDS aperture
// keeps the workgroup-relative offset in addr[31:0]).
DEV unsigned lds32(const void* p) { return (unsigned)(size_t)p; }

DEV void async_ld_f32(unsigned lds_off, const float* gaddr) {
    asm volatile("global_load_async_to_lds_b32 %0, %1, off"
                 :: "v"(lds_off), "v"(gaddr)
                 : "memory");
}
DEV void wait_async() {
    asm volatile("s_wait_asynccnt 0x0" ::: "memory");
}

// Load a 16-element bf16 WMMA fragment from a row-major row pointer.
// Layout per CDNA5 ISA 7.12.2 (16-bit A 16x32): for lane half = lane/16,
// vector element pair (2v,2v+1) holds K = 8*half+2v (v<4) / 16+8*half+2(v-4).
// The B fragment (32x16, lane = N column) uses the same K striping.
DEV v16bf load_frag(const bf16* rowptr, int k0, int half) {
    v16bf f;
#pragma unroll
    for (int v = 0; v < 8; ++v) {
        int k = k0 + ((v < 4) ? (8 * half + 2 * v) : (16 + 8 * half + 2 * (v - 4)));
        f[2 * v]     = rowptr[k];
        f[2 * v + 1] = rowptr[k + 1];
    }
    return f;
}

// ---------------------------------------------------------------------------
// Utility: f32 -> bf16 array convert (weights)
// ---------------------------------------------------------------------------
__global__ void k_f32_to_bf16(const float* __restrict__ src, bf16* __restrict__ dst, int n) {
    int i = blockIdx.x * blockDim.x + threadIdx.x;
    if (i < n) dst[i] = f2bf(src[i]);
}

// ---------------------------------------------------------------------------
// LayerNorm over channels. x is (B, CH, L); xn_bf is (B*L, CH) bf16.
// One wave per (b,l) row; lane covers 3 channels (96 = 3*32).
// ---------------------------------------------------------------------------
__global__ void k_layernorm(const float* __restrict__ x,
                            const float* __restrict__ g,
                            const float* __restrict__ be,
                            bf16* __restrict__ xn) {
    int tid  = threadIdx.x;
    int lane = tid & 31;
    int row  = blockIdx.x * 8 + (tid >> 5);       // b*L + l
    int b = row >> 14, l = row & (kL - 1);
    const float* xp = x + (size_t)b * kCH * kL + l;
    float v0 = xp[(size_t)(lane)      * kL];
    float v1 = xp[(size_t)(lane + 32) * kL];
    float v2 = xp[(size_t)(lane + 64) * kL];
    float s  = v0 + v1 + v2;
    float ss = v0 * v0 + v1 * v1 + v2 * v2;
#pragma unroll
    for (int m = 16; m; m >>= 1) {
        s  += __shfl_xor(s,  m, 32);
        ss += __shfl_xor(ss, m, 32);
    }
    float mean = s * (1.f / kCH);
    float var  = ss * (1.f / kCH) - mean * mean;
    float inv  = rsqrtf(var + 1e-5f);
    bf16* o = xn + (size_t)row * kCH;
    o[lane]      = f2bf((v0 - mean) * inv * g[lane]      + be[lane]);
    o[lane + 32] = f2bf((v1 - mean) * inv * g[lane + 32] + be[lane + 32]);
    o[lane + 64] = f2bf((v2 - mean) * inv * g[lane + 64] + be[lane + 64]);
}

// ---------------------------------------------------------------------------
// Input projection GEMM: xz[NROW,384] = xn[NROW,96] @ in_w[384,96]^T.
// Register-blocked: each wave owns one 16-row M-tile and a group of 4 N-tiles
// (64 cols); A fragments (K=96 -> 3) are loaded once and reused 4x.
// ng 0..2 -> xi cols, ng 3..5 -> z cols (wave-uniform scalar branch).
// ---------------------------------------------------------------------------
__global__ void k_gemm_in(const bf16* __restrict__ xn,
                          const bf16* __restrict__ w,     // [384,96] bf16
                          float* __restrict__ xi,
                          float* __restrict__ z,
                          int reverse) {
    int tid = threadIdx.x, wave = tid >> 5, lane = tid & 31;
    int t  = blockIdx.x * 8 + wave;                // wave id
    int tm = t / 6;
    int ng = __builtin_amdgcn_readfirstlane(t % 6);
    int half = lane >> 4, m16 = lane & 15;

    int arow = tm * 16 + m16;                      // direction-space row
    int b = arow >> 14, p = arow & (kL - 1);
    int src = reverse ? (b * kL + (kL - 1 - p)) : arow;
    const bf16* aptr = xn + (size_t)src * kCH;

    v16bf afr[3];
#pragma unroll
    for (int kk = 0; kk < 3; ++kk) afr[kk] = load_frag(aptr, kk * 32, half);

    v8f acc[4] = {};
#pragma unroll
    for (int j = 0; j < 4; ++j) {
        const bf16* bptr = w + (size_t)((ng * 4 + j) * 16 + m16) * kCH;
#pragma unroll
        for (int kk = 0; kk < 3; ++kk) {
            v16bf bb = load_frag(bptr, kk * 32, half);
            acc[j] = __builtin_amdgcn_wmma_f32_16x16x32_bf16(false, afr[kk], false, bb,
                                                             (short)0, acc[j], false, false);
        }
    }

    float* dst     = (ng < 3) ? xi : z;
    int    colbase = (ng < 3) ? ng * 64 : (ng - 3) * 64;
#pragma unroll
    for (int j = 0; j < 4; ++j) {
        int col = colbase + j * 16 + m16;
#pragma unroll
        for (int v = 0; v < 8; ++v) {
            int row = tm * 16 + v + 8 * half;
            dst[(size_t)row * kDIN + col] = acc[j][v];
        }
    }
}

// ---------------------------------------------------------------------------
// Causal depthwise conv (DCONV=4) + SiLU, per direction-space buffer.
// ---------------------------------------------------------------------------
__global__ void k_conv(const float* __restrict__ xi,
                       const float* __restrict__ cw,   // [192,1,4]
                       const float* __restrict__ cb,
                       float* __restrict__ xc) {
    int i = blockIdx.x * blockDim.x + threadIdx.x;     // over NROW*192
    int r = i / kDIN, d = i - r * kDIN;
    int b = r >> 14, p = r & (kL - 1);
    float acc = cb[d];
#pragma unroll
    for (int j = 0; j < 4; ++j) {
        int pp = p - 3 + j;
        if (pp >= 0) acc += cw[d * 4 + j] * xi[(size_t)(b * kL + pp) * kDIN + d];
    }
    xc[i] = silu(acc);
}

// ---------------------------------------------------------------------------
// x-projection (38 outputs, K=192) + softplus(delta).  One row per block,
// 192 threads; xc row staged in LDS.
// ---------------------------------------------------------------------------
__global__ void k_proj(const float* __restrict__ xc,
                       const float* __restrict__ xproj,  // [38,192]
                       const float* __restrict__ dtw,    // [192,6]
                       const float* __restrict__ dtb,
                       float* __restrict__ Bm,           // [NROW,16]
                       float* __restrict__ Cm,           // [NROW,16]
                       float* __restrict__ delta) {      // [NROW,192]
    __shared__ float sx[kDIN];
    __shared__ float sdt[kDTR];
    int r = blockIdx.x, tid = threadIdx.x;
    sx[tid] = xc[(size_t)r * kDIN + tid];
    __syncthreads();
    if (tid < kDBL) {
        const float* wrow = xproj + tid * kDIN;
        float acc = 0.f;
#pragma unroll 8
        for (int k = 0; k < kDIN; ++k) acc += sx[k] * wrow[k];
        if (tid < kDTR)               sdt[tid] = acc;
        else if (tid < kDTR + 16)     Bm[(size_t)r * 16 + (tid - kDTR)] = acc;
        else                          Cm[(size_t)r * 16 + (tid - kDTR - 16)] = acc;
    }
    __syncthreads();
    float acc = dtb[tid];
#pragma unroll
    for (int j = 0; j < kDTR; ++j) acc += sdt[j] * dtw[tid * kDTR + j];
    float sp = (acc > 20.f) ? acc : log1pf(__expf(acc));
    delta[(size_t)r * kDIN + tid] = sp;
}

// ---------------------------------------------------------------------------
// Chunked selective scan.  Recurrence h = a*h + u, a = exp(delta*A),
// u = delta*x*B.  Pass1: per-chunk product P and zero-start partial q.
// Pass2: sequential prefix over chunks.  Pass3: replay with corrected h0,
// reduce over states (shfl width 16), gate with SiLU(z).
// Block = 256 threads = 16 channels x 16 states.  Per-16-step operand tiles
// are streamed into double-buffered LDS with GLOBAL_LOAD_ASYNC_TO_LDS_B32,
// overlapping global fetch of tile t+1 with the recurrence steps of tile t.
// ---------------------------------------------------------------------------
__global__ void k_scan_pass1(const float* __restrict__ delta,
                             const float* __restrict__ xc,
                             const float* __restrict__ Bm,
                             const float* __restrict__ Alog,
                             float* __restrict__ cA,
                             float* __restrict__ cH) {
    __shared__ float sde[2][kTS * 16];
    __shared__ float sxv[2][kTS * 16];
    __shared__ float sB [2][kTS * 16];
    int tid = threadIdx.x;
    int chunk = blockIdx.x / 12, dg = blockIdx.x % 12;
    int b = blockIdx.y;
    int di = tid >> 4, n = tid & 15;
    int d = dg * 16 + di;
    float A = -__expf(Alog[d * 16 + n]);
    float P = 1.f, q = 0.f;
    size_t base = (size_t)(b * kL + chunk * kCL);
    int el = di * 16 + n;                 // this thread's tile element

    auto issue = [&](int t0, int pb) {
        size_t r = base + t0 + di;        // thread loads step di, elem n
        async_ld_f32(lds32(&sde[pb][el]), delta + r * kDIN + dg * 16 + n);
        async_ld_f32(lds32(&sxv[pb][el]), xc    + r * kDIN + dg * 16 + n);
        async_ld_f32(lds32(&sB [pb][el]), Bm    + r * 16 + n);
    };

    issue(0, 0);
    wait_async();
    __syncthreads();
    for (int tt = 0; tt < kNT; ++tt) {
        if (tt + 1 < kNT) issue((tt + 1) * kTS, (tt + 1) & 1);
        int pb = tt & 1;
#pragma unroll
        for (int s = 0; s < kTS; ++s) {
            float de = sde[pb][s * 16 + di];
            float a  = __expf(de * A);
            float u  = de * sxv[pb][s * 16 + di] * sB[pb][s * 16 + n];
            P *= a;
            q  = a * q + u;
        }
        wait_async();
        __syncthreads();
    }
    size_t idx = (((size_t)chunk * kB + b) * kDIN + d) * 16 + n;
    cA[idx] = P;
    cH[idx] = q;
}

__global__ void k_scan_pass2(const float* __restrict__ cA,
                             const float* __restrict__ cH,
                             float* __restrict__ hstart) {
    int tid = threadIdx.x;
    int dg = blockIdx.x, b = blockIdx.y;
    int d = dg * 16 + (tid >> 4), n = tid & 15;
    float h = 0.f;
    for (int c = 0; c < kNC; ++c) {
        size_t idx = (((size_t)c * kB + b) * kDIN + d) * 16 + n;
        float P = cA[idx], q = cH[idx];
        hstart[idx] = h;
        h = P * h + q;
    }
}

__global__ void k_scan_pass3(const float* __restrict__ delta,
                             const float* __restrict__ xc,
                             const float* __restrict__ Bm,
                             const float* __restrict__ Cm,
                             const float* __restrict__ z,
                             const float* __restrict__ Alog,
                             const float* __restrict__ Dvec,
                             const float* __restrict__ hstart,
                             float* __restrict__ yg) {
    __shared__ float sde[2][kTS * 16];
    __shared__ float sxv[2][kTS * 16];
    __shared__ float sB [2][kTS * 16];
    __shared__ float sC [2][kTS * 16];
    __shared__ float sZ [2][kTS * 16];
    int tid = threadIdx.x;
    int chunk = blockIdx.x / 12, dg = blockIdx.x % 12;
    int b = blockIdx.y;
    int di = tid >> 4, n = tid & 15;
    int d = dg * 16 + di;
    float A  = -__expf(Alog[d * 16 + n]);
    float Dd = Dvec[d];
    size_t idx = (((size_t)chunk * kB + b) * kDIN + d) * 16 + n;
    float h = hstart[idx];
    size_t base = (size_t)(b * kL + chunk * kCL);
    int el = di * 16 + n;

    auto issue = [&](int t0, int pb) {
        size_t r = base + t0 + di;
        async_ld_f32(lds32(&sde[pb][el]), delta + r * kDIN + dg * 16 + n);
        async_ld_f32(lds32(&sxv[pb][el]), xc    + r * kDIN + dg * 16 + n);
        async_ld_f32(lds32(&sZ [pb][el]), z     + r * kDIN + dg * 16 + n);
        async_ld_f32(lds32(&sB [pb][el]), Bm    + r * 16 + n);
        async_ld_f32(lds32(&sC [pb][el]), Cm    + r * 16 + n);
    };

    issue(0, 0);
    wait_async();
    __syncthreads();
    for (int tt = 0; tt < kNT; ++tt) {
        if (tt + 1 < kNT) issue((tt + 1) * kTS, (tt + 1) & 1);
        int pb = tt & 1;
#pragma unroll
        for (int s = 0; s < kTS; ++s) {
            float de = sde[pb][s * 16 + di];
            float xv = sxv[pb][s * 16 + di];
            float a  = __expf(de * A);
            h = a * h + de * xv * sB[pb][s * 16 + n];
            float yc = h * sC[pb][s * 16 + n];
            yc += __shfl_xor(yc, 8, 16);
            yc += __shfl_xor(yc, 4, 16);
            yc += __shfl_xor(yc, 2, 16);
            yc += __shfl_xor(yc, 1, 16);
            if (n == 0) {
                float zv = sZ[pb][s * 16 + di];
                yg[(base + tt * kTS + s) * kDIN + d] = (yc + xv * Dd) * silu(zv);
            }
        }
        wait_async();
        __syncthreads();
    }
}

// ---------------------------------------------------------------------------
// Combine directions into bf16 (backward re-reversed to forward order).
// ---------------------------------------------------------------------------
__global__ void k_combine(const float* __restrict__ ygf,
                          const float* __restrict__ ygb,
                          bf16* __restrict__ ycomb) {
    int i = blockIdx.x * blockDim.x + threadIdx.x;     // over NROW*192
    int r = i / kDIN, d = i - r * kDIN;
    int b = r >> 14, l = r & (kL - 1);
    size_t rb = (size_t)(b * kL + (kL - 1 - l));
    ycomb[i] = f2bf(ygf[i] + ygb[rb * kDIN + d]);
}

// ---------------------------------------------------------------------------
// Output projection GEMM + residual: out[b,c,l] = x[b,c,l] + ycomb[b,l,:]@out_w[c,:]
// Register-blocked: each wave owns one M-tile and 3 N-tiles; the 6 K-step A
// fragments are loaded once and reused 3x (18 WMMAs per wave).
// ---------------------------------------------------------------------------
__global__ void k_gemm_out(const bf16* __restrict__ ycomb,  // [NROW,192]
                           const bf16* __restrict__ w,      // [96,192]
                           const float* __restrict__ x,
                           float* __restrict__ out) {
    int tid = threadIdx.x, wave = tid >> 5, lane = tid & 31;
    int t  = blockIdx.x * 8 + wave;
    int tm = t / 2;
    int ng = __builtin_amdgcn_readfirstlane(t % 2);    // 3 N-tiles per group
    int half = lane >> 4, m16 = lane & 15;

    const bf16* aptr = ycomb + (size_t)(tm * 16 + m16) * kDIN;
    v16bf afr[6];
#pragma unroll
    for (int kk = 0; kk < 6; ++kk) afr[kk] = load_frag(aptr, kk * 32, half);

    v8f acc[3] = {};
#pragma unroll
    for (int j = 0; j < 3; ++j) {
        const bf16* bptr = w + (size_t)((ng * 3 + j) * 16 + m16) * kDIN;
#pragma unroll
        for (int kk = 0; kk < 6; ++kk) {
            v16bf bb = load_frag(bptr, kk * 32, half);
            acc[j] = __builtin_amdgcn_wmma_f32_16x16x32_bf16(false, afr[kk], false, bb,
                                                             (short)0, acc[j], false, false);
        }
    }
#pragma unroll
    for (int j = 0; j < 3; ++j) {
        int c = (ng * 3 + j) * 16 + m16;
#pragma unroll
        for (int v = 0; v < 8; ++v) {
            int row = tm * 16 + v + 8 * half;
            int b = row >> 14, l = row & (kL - 1);
            size_t o = (size_t)b * kCH * kL + (size_t)c * kL + l;
            out[o] = x[o] + acc[j][v];
        }
    }
}

// ---------------------------------------------------------------------------
// Host launcher
// ---------------------------------------------------------------------------
static inline size_t align256(size_t x) { return (x + 255) & ~(size_t)255; }

extern "C" void kernel_launch(void* const* d_in, const int* in_sizes, int n_in,
                              void* d_out, int out_size, void* d_ws, size_t ws_size,
                              hipStream_t stream) {
    (void)in_sizes; (void)n_in; (void)out_size; (void)ws_size;
    const float* x     = (const float*)d_in[0];
    const float* ln_g  = (const float*)d_in[1];
    const float* ln_b  = (const float*)d_in[2];
    const float* out_w = (const float*)d_in[3];
    const float* in_w[2]    = {(const float*)d_in[4],  (const float*)d_in[12]};
    const float* conv_w[2]  = {(const float*)d_in[5],  (const float*)d_in[13]};
    const float* conv_b[2]  = {(const float*)d_in[6],  (const float*)d_in[14]};
    const float* xproj_w[2] = {(const float*)d_in[7],  (const float*)d_in[15]};
    const float* dt_w[2]    = {(const float*)d_in[8],  (const float*)d_in[16]};
    const float* dt_b[2]    = {(const float*)d_in[9],  (const float*)d_in[17]};
    const float* A_log[2]   = {(const float*)d_in[10], (const float*)d_in[18]};
    const float* Dv[2]      = {(const float*)d_in[11], (const float*)d_in[19]};
    float* out = (float*)d_out;

    // ---- workspace bump allocation ----
    char* wsp = (char*)d_ws;
    size_t off = 0;
    auto alloc = [&](size_t bytes) -> char* {
        char* p = wsp + off;
        off += align256(bytes);
        return p;
    };
    const size_t fdin  = (size_t)kNROW * kDIN * sizeof(float);   // 25.2 MB
    const size_t fst   = (size_t)kNROW * 16   * sizeof(float);   //  2.1 MB
    const size_t fchnk = (size_t)kNC * kB * kDIN * 16 * sizeof(float);

    bf16* xn_bf    = (bf16*)alloc((size_t)kNROW * kCH * sizeof(bf16));
    bf16* inw_bf[2]= {(bf16*)alloc(2 * kDIN * kCH * sizeof(bf16)),
                      (bf16*)alloc(2 * kDIN * kCH * sizeof(bf16))};
    bf16* outw_bf  = (bf16*)alloc((size_t)kCH * kDIN * sizeof(bf16));
    float* xi[2]    = {(float*)alloc(fdin), (float*)alloc(fdin)};
    float* zb[2]    = {(float*)alloc(fdin), (float*)alloc(fdin)};
    float* xc[2]    = {(float*)alloc(fdin), (float*)alloc(fdin)};
    float* delta[2] = {(float*)alloc(fdin), (float*)alloc(fdin)};
    float* Bm[2]    = {(float*)alloc(fst),  (float*)alloc(fst)};
    float* Cm[2]    = {(float*)alloc(fst),  (float*)alloc(fst)};
    float* cA[2]    = {(float*)alloc(fchnk), (float*)alloc(fchnk)};
    float* cH[2]    = {(float*)alloc(fchnk), (float*)alloc(fchnk)};
    float* hst[2]   = {(float*)alloc(fchnk), (float*)alloc(fchnk)};
    bf16* ycomb    = (bf16*)alloc((size_t)kNROW * kDIN * sizeof(bf16));
    float* yg[2]    = {xi[0], xi[1]};   // xi is dead after conv; reuse for gated y

    // ---- weight conversion ----
    k_f32_to_bf16<<<(2 * kDIN * kCH + 255) / 256, 256, 0, stream>>>(in_w[0], inw_bf[0], 2 * kDIN * kCH);
    k_f32_to_bf16<<<(2 * kDIN * kCH + 255) / 256, 256, 0, stream>>>(in_w[1], inw_bf[1], 2 * kDIN * kCH);
    k_f32_to_bf16<<<(kCH * kDIN + 255) / 256, 256, 0, stream>>>(out_w, outw_bf, kCH * kDIN);

    // ---- layernorm ----
    k_layernorm<<<kNROW / 8, 256, 0, stream>>>(x, ln_g, ln_b, xn_bf);

    // ---- input projection (WMMA), conv+SiLU, x-proj+softplus ----
    const int g1 = (kNROW / 16) * 6 / 8;                         // 1536 blocks
    for (int dir = 0; dir < 2; ++dir) {
        k_gemm_in<<<g1, 256, 0, stream>>>(xn_bf, inw_bf[dir], xi[dir], zb[dir], dir);
        k_conv<<<(kNROW * kDIN) / 256, 256, 0, stream>>>(xi[dir], conv_w[dir], conv_b[dir], xc[dir]);
        k_proj<<<kNROW, kDIN, 0, stream>>>(xc[dir], xproj_w[dir], dt_w[dir], dt_b[dir],
                                           Bm[dir], Cm[dir], delta[dir]);
    }

    // ---- chunk-parallel selective scan (async double-buffered LDS) ----
    for (int dir = 0; dir < 2; ++dir) {
        k_scan_pass1<<<dim3(12 * kNC, kB), 256, 0, stream>>>(delta[dir], xc[dir], Bm[dir],
                                                             A_log[dir], cA[dir], cH[dir]);
        k_scan_pass2<<<dim3(12, kB), 256, 0, stream>>>(cA[dir], cH[dir], hst[dir]);
        k_scan_pass3<<<dim3(12 * kNC, kB), 256, 0, stream>>>(delta[dir], xc[dir], Bm[dir], Cm[dir],
                                                             zb[dir], A_log[dir], Dv[dir],
                                                             hst[dir], yg[dir]);
    }

    // ---- combine directions, output projection (WMMA) + residual ----
    k_combine<<<(kNROW * kDIN) / 256, 256, 0, stream>>>(yg[0], yg[1], ycomb);
    const int g2 = (kNROW / 16) * 2 / 8;                         // 512 blocks
    k_gemm_out<<<g2, 256, 0, stream>>>(ycomb, outw_bf, x, out);
}
